// Attention_45148696216598
// MI455X (gfx1250) — matrix operand order repaired and checked by reading, not compiled, not run on hardware
//
#include <hip/hip_runtime.h>
#include <hip/hip_bf16.h>

// ---------------------------------------------------------------------------
// GQA attention for MI455X (gfx1250, wave32, WMMA).
// B=2, N=2048, DIM=1024, H=16, KVH=4, DH=64, SCALE=0.125
// All GEMM work via v_wmma_f32_16x16x32_bf16 (f32 accum), flash-style softmax.
// GEMMs register-block 32x32 per wave (2x2 WMMA tiles) -> 2 b128 loads/WMMA.
// Prefetch 16 K-steps (512 elems) ahead in GEMMs, next 32-key panel in attn, at WGP scope (locality 3).
// ---------------------------------------------------------------------------

typedef __attribute__((ext_vector_type(16))) __bf16 v16bf;
typedef __attribute__((ext_vector_type(8)))  __bf16 v8bf;
typedef __attribute__((ext_vector_type(8)))  float  v8f;

__device__ __forceinline__ unsigned short f2bf(float f) {
    unsigned int u = __float_as_uint(f);
    u += 0x7FFFu + ((u >> 16) & 1u);          // round-to-nearest-even
    return (unsigned short)(u >> 16);
}

// A-fragment: 16 halves per lane, one contiguous group of 16 starting at p0.
__device__ __forceinline__ v16bf load_a_frag(const unsigned short* p0,
                                             const unsigned short* p1) {
    v8bf lo = *(const v8bf*)p0;
    v8bf hi = *(const v8bf*)(p0 + 8);
    v16bf a;
#pragma unroll
    for (int i = 0; i < 8; ++i) { a[i] = lo[i]; a[i + 8] = hi[i]; }
    return a;
}

__device__ __forceinline__ v8f wmma_bf16(v16bf a, v16bf b, v8f c) {
    return __builtin_amdgcn_wmma_f32_16x16x32_bf16(
        /*neg_a=*/false, a, /*neg_b=*/false, b,
        /*c_mod=*/(short)0, c, /*reuse_a=*/false, /*reuse_b=*/false);
}

// ------------------------- precision-convert prologue ----------------------

__global__ void k_cvt_x(const float* __restrict__ x, unsigned short* __restrict__ xb) {
    int i = blockIdx.x * blockDim.x + threadIdx.x;   // 4096*1024 threads
    xb[i] = f2bf(x[i]);
}

// wqkvT[o][i] rows: o<1024 -> Wq col o; o<1280 -> Wk col; else Wv col.
__global__ void k_pack_wqkv(const float* __restrict__ Wq, const float* __restrict__ Wk,
                            const float* __restrict__ Wv, unsigned short* __restrict__ wT) {
    int i = blockIdx.x * blockDim.x + threadIdx.x;   // 1536*1024 threads
    int o = i >> 10, ii = i & 1023;
    float w;
    if (o < 1024)      w = Wq[ii * 1024 + o];
    else if (o < 1280) w = Wk[ii * 256 + (o - 1024)];
    else               w = Wv[ii * 256 + (o - 1280)];
    wT[(size_t)o * 1024 + ii] = f2bf(w);
}

__global__ void k_pack_wo(const float* __restrict__ Wo, unsigned short* __restrict__ wT) {
    int i = blockIdx.x * blockDim.x + threadIdx.x;   // 1024*1024 threads
    int o = i >> 10, ii = i & 1023;
    wT[(size_t)o * 1024 + ii] = f2bf(Wo[ii * 1024 + o]);
}

// ------------------------- fused QKV projection GEMM -----------------------
// C[4096][1536] = xb[4096][1024] @ Wqkv ; 32x32 output per wave (2x2 tiles).
// Q scaled by 0.125 and stored [b][h][n][dh]; K stored [b][kvh][n][dh];
// V stored transposed [b][kvh][dh][n] (directly a WMMA B^T buffer).

__device__ __forceinline__ void qkv_store(float v, int row, int col,
                                          unsigned short* __restrict__ Qb,
                                          unsigned short* __restrict__ Kb,
                                          unsigned short* __restrict__ VbT) {
    int bb = row >> 11, n = row & 2047;
    if (col < 1024) {                                 // Q (pre-scaled)
        int h = col >> 6, dh = col & 63;
        Qb[(((bb * 16 + h) * 2048) + n) * 64 + dh] = f2bf(v * 0.125f);
    } else if (col < 1280) {                          // K
        int c2 = col - 1024, kvh = c2 >> 6, dh = c2 & 63;
        Kb[(((bb * 4 + kvh) * 2048) + n) * 64 + dh] = f2bf(v);
    } else {                                          // V, transposed
        int c2 = col - 1280, kvh = c2 >> 6, dh = c2 & 63;
        VbT[((bb * 4 + kvh) * 64 + dh) * 2048 + n] = f2bf(v);
    }
}

__global__ __launch_bounds__(256) void k_gemm_qkv(
    const unsigned short* __restrict__ xb, const unsigned short* __restrict__ wT,
    unsigned short* __restrict__ Qb, unsigned short* __restrict__ Kb,
    unsigned short* __restrict__ VbT) {
    int wave = (blockIdx.x * 256 + threadIdx.x) >> 5;
    int lane = threadIdx.x & 31, l15 = lane & 15, hi = lane >> 4, kOff = hi * 16;
    int tm = wave / 48, tn = wave % 48;               // 128 x 48 macro-tiles (32x32)
    const unsigned short* arow0 = xb + (size_t)(tm * 32 + l15) * 1024;
    const unsigned short* arow1 = arow0 + 16 * 1024;
    const unsigned short* brow0 = wT + (size_t)(tn * 32 + l15) * 1024 + hi * 16;
    const unsigned short* brow1 = brow0 + 16 * 1024;
    v8f c00 = {}, c01 = {}, c10 = {}, c11 = {};
    for (int k = 0; k < 1024; k += 32) {
        __builtin_prefetch(arow0 + k + 512, 0, 3);    // WGP-scope prefetch, 16 K-steps (512 elems) ahead
        __builtin_prefetch(brow0 + k + 512, 0, 3);
        v16bf a0 = load_a_frag(arow0 + k + kOff, arow0 + k + kOff + 16);
        v16bf a1 = load_a_frag(arow1 + k + kOff, arow1 + k + kOff + 16);
        v16bf b0 = *(const v16bf*)(brow0 + k);
        v16bf b1 = *(const v16bf*)(brow1 + k);
        c00 = wmma_bf16(a0, b0, c00);
        c01 = wmma_bf16(a0, b1, c01);
        c10 = wmma_bf16(a1, b0, c10);
        c11 = wmma_bf16(a1, b1, c11);
    }
    int col0 = tn * 32 + l15, col1 = col0 + 16;
#pragma unroll
    for (int r = 0; r < 8; ++r) {
        int row0 = tm * 32 + r + hi * 8, row1 = row0 + 16;
        qkv_store(c00[r], row0, col0, Qb, Kb, VbT);
        qkv_store(c01[r], row0, col1, Qb, Kb, VbT);
        qkv_store(c10[r], row1, col0, Qb, Kb, VbT);
        qkv_store(c11[r], row1, col1, Qb, Kb, VbT);
    }
}

// ------------------------- flash attention ---------------------------------
// One wave per (b, h, 16-row query tile). 32 keys per step:
//   S(16x32) = Q(16x64) K^T  (4 WMMAs) -> mask -> online softmax ->
//   P staged via LDS (C-layout -> A-layout) -> O(16x64) += P V (4 WMMAs).

__global__ __launch_bounds__(256) void k_attn(
    const unsigned short* __restrict__ Qb, const unsigned short* __restrict__ Kb,
    const unsigned short* __restrict__ VbT, const unsigned char* __restrict__ mask,
    unsigned short* __restrict__ Ob) {
    __shared__ unsigned short plds[8][512];           // 16x32 bf16 per wave
    int wave = (blockIdx.x * 256 + threadIdx.x) >> 5;
    int wid  = threadIdx.x >> 5;
    int lane = threadIdx.x & 31, l15 = lane & 15, hi = lane >> 4, kOff = hi * 16;
    int qt = wave & 127, h = (wave >> 7) & 15, b = wave >> 11;
    int kvh = h >> 2;                                 // GQA: 4 q-heads per kv-head

    const unsigned short* Qbase = Qb + (size_t)((b * 16 + h) * 2048 + qt * 16) * 64;
    const unsigned short* Kbase = Kb + (size_t)((b * 4 + kvh) * 2048) * 64;
    const unsigned short* VT    = VbT + (size_t)((b * 4 + kvh) * 64) * 2048;
    unsigned short* pbuf = &plds[wid][0];

    const unsigned short* qrow = Qbase + l15 * 64;
    v16bf aq0 = load_a_frag(qrow + kOff,      qrow + kOff + 16);        // dh 0..31
    v16bf aq1 = load_a_frag(qrow + 32 + kOff, qrow + 32 + kOff + 16);   // dh 32..63

    v8f o0 = {}, o1 = {}, o2 = {}, o3 = {};
    float m[8], l[8];
#pragma unroll
    for (int r = 0; r < 8; ++r) { m[r] = -3.0e38f; l[r] = 0.0f; }

    for (int kt = 0; kt < 64; ++kt) {
        int j0 = kt * 32, j1 = j0 + 16;
        const unsigned short* k0row = Kbase + (size_t)(j0 + l15) * 64 + hi * 16;
        const unsigned short* k1row = Kbase + (size_t)(j1 + l15) * 64 + hi * 16;
        __builtin_prefetch(k0row + 2048, 0, 3);       // next 32-key panel, WGP scope
        v8f s0 = {}, s1 = {};
        s0 = wmma_bf16(aq0, *(const v16bf*)(k0row),      s0);
        s0 = wmma_bf16(aq1, *(const v16bf*)(k0row + 32), s0);
        s1 = wmma_bf16(aq0, *(const v16bf*)(k1row),      s1);
        s1 = wmma_bf16(aq1, *(const v16bf*)(k1row + 32), s1);

#pragma unroll
        for (int r = 0; r < 8; ++r) {
            int rowg = qt * 16 + r + hi * 8;
            const unsigned char* mrow = mask + (size_t)rowg * 2048;
            float v0 = mrow[j0 + l15] ? -1e9f : s0[r];
            float v1 = mrow[j1 + l15] ? -1e9f : s1[r];
            float tmax = fmaxf(v0, v1);
#pragma unroll
            for (int off = 1; off < 16; off <<= 1)
                tmax = fmaxf(tmax, __shfl_xor(tmax, off, 32));
            float mnew = fmaxf(m[r], tmax);
            float corr = __expf(m[r] - mnew);
            float p0 = __expf(v0 - mnew);
            float p1 = __expf(v1 - mnew);
            float ps = p0 + p1;
#pragma unroll
            for (int off = 1; off < 16; off <<= 1)
                ps += __shfl_xor(ps, off, 32);
            l[r] = l[r] * corr + ps;
            m[r] = mnew;
            o0[r] *= corr; o1[r] *= corr; o2[r] *= corr; o3[r] *= corr;
            pbuf[(r + hi * 8) * 32 + l15]      = f2bf(p0);
            pbuf[(r + hi * 8) * 32 + l15 + 16] = f2bf(p1);
        }
        __syncthreads();
        const unsigned short* prow = pbuf + l15 * 32;
        v16bf pa = load_a_frag(prow + kOff, prow + kOff + 16);
        const unsigned short* vcol = VT + (size_t)l15 * 2048 + j0 + hi * 16;
        o0 = wmma_bf16(pa, *(const v16bf*)(vcol),             o0);
        o1 = wmma_bf16(pa, *(const v16bf*)(vcol + 16 * 2048), o1);
        o2 = wmma_bf16(pa, *(const v16bf*)(vcol + 32 * 2048), o2);
        o3 = wmma_bf16(pa, *(const v16bf*)(vcol + 48 * 2048), o3);
        __syncthreads();
    }

#pragma unroll
    for (int r = 0; r < 8; ++r) {
        float inv = 1.0f / l[r];
        int n = qt * 16 + r + hi * 8;
        unsigned short* orow = Ob + (size_t)(b * 2048 + n) * 1024 + h * 64 + l15;
        orow[0]  = f2bf(o0[r] * inv);
        orow[16] = f2bf(o1[r] * inv);
        orow[32] = f2bf(o2[r] * inv);
        orow[48] = f2bf(o3[r] * inv);
    }
}

// ------------------------- output projection -------------------------------
// out[4096][1024] = Ob @ Wo + bo, f32 result. 32x32 per wave (2x2 tiles).

__global__ __launch_bounds__(256) void k_gemm_out(
    const unsigned short* __restrict__ Ob, const unsigned short* __restrict__ woT,
    const float* __restrict__ bo, float* __restrict__ out) {
    int wave = (blockIdx.x * 256 + threadIdx.x) >> 5;
    int lane = threadIdx.x & 31, l15 = lane & 15, hi = lane >> 4, kOff = hi * 16;
    int tm = wave >> 5, tn = wave & 31;               // 128 x 32 macro-tiles (32x32)
    const unsigned short* arow0 = Ob  + (size_t)(tm * 32 + l15) * 1024;
    const unsigned short* arow1 = arow0 + 16 * 1024;
    const unsigned short* brow0 = woT + (size_t)(tn * 32 + l15) * 1024 + hi * 16;
    const unsigned short* brow1 = brow0 + 16 * 1024;
    v8f c00 = {}, c01 = {}, c10 = {}, c11 = {};
    for (int k = 0; k < 1024; k += 32) {
        __builtin_prefetch(arow0 + k + 512, 0, 3);
        __builtin_prefetch(brow0 + k + 512, 0, 3);
        v16bf a0 = load_a_frag(arow0 + k + kOff, arow0 + k + kOff + 16);
        v16bf a1 = load_a_frag(arow1 + k + kOff, arow1 + k + kOff + 16);
        v16bf b0 = *(const v16bf*)(brow0 + k);
        v16bf b1 = *(const v16bf*)(brow1 + k);
        c00 = wmma_bf16(a0, b0, c00);
        c01 = wmma_bf16(a0, b1, c01);
        c10 = wmma_bf16(a1, b0, c10);
        c11 = wmma_bf16(a1, b1, c11);
    }
    int col0 = tn * 32 + l15, col1 = col0 + 16;
    float bias0 = bo[col0], bias1 = bo[col1];
#pragma unroll
    for (int r = 0; r < 8; ++r) {
        int row0 = tm * 32 + r + hi * 8, row1 = row0 + 16;
        out[(size_t)row0 * 1024 + col0] = c00[r] + bias0;
        out[(size_t)row0 * 1024 + col1] = c01[r] + bias1;
        out[(size_t)row1 * 1024 + col0] = c10[r] + bias0;
        out[(size_t)row1 * 1024 + col1] = c11[r] + bias1;
    }
}

// ------------------------- host launcher -----------------------------------

extern "C" void kernel_launch(void* const* d_in, const int* in_sizes, int n_in,
                              void* d_out, int out_size, void* d_ws, size_t ws_size,
                              hipStream_t stream) {
    (void)in_sizes; (void)n_in; (void)out_size; (void)ws_size;
    const float*         x    = (const float*)d_in[0];
    const unsigned char* mask = (const unsigned char*)d_in[1];  // jnp.bool_ = 1 byte
    const float*         Wq   = (const float*)d_in[2];
    const float*         Wk   = (const float*)d_in[3];
    const float*         Wv   = (const float*)d_in[4];
    const float*         Wo   = (const float*)d_in[5];
    const float*         bo   = (const float*)d_in[6];
    float*               out  = (float*)d_out;

    // Workspace layout (bf16 elements, ~34.6 MB total)
    unsigned short* ws    = (unsigned short*)d_ws;
    unsigned short* xb    = ws;                       // 4096*1024
    unsigned short* wqkvT = xb    + (size_t)4096 * 1024;
    unsigned short* woT   = wqkvT + (size_t)1536 * 1024;
    unsigned short* Qb    = woT   + (size_t)1024 * 1024;
    unsigned short* Kb    = Qb    + (size_t)2 * 16 * 2048 * 64;
    unsigned short* VbT   = Kb    + (size_t)2 * 4 * 2048 * 64;
    unsigned short* Ob    = VbT   + (size_t)2 * 4 * 2048 * 64;

    k_cvt_x    <<<(4096 * 1024) / 256, 256, 0, stream>>>(x, xb);
    k_pack_wqkv<<<(1536 * 1024) / 256, 256, 0, stream>>>(Wq, Wk, Wv, wqkvT);
    k_pack_wo  <<<(1024 * 1024) / 256, 256, 0, stream>>>(Wo, woT);
    k_gemm_qkv <<<768,  256, 0, stream>>>(xb, wqkvT, Qb, Kb, VbT);   // 6144 waves
    k_attn     <<<512,  256, 0, stream>>>(Qb, Kb, VbT, mask, Ob);    // 4096 waves
    k_gemm_out <<<512,  256, 0, stream>>>(Ob, woT, bo, out);         // 4096 waves
}
